// MultiHeadAttention_15144054686315
// MI455X (gfx1250) — matrix operand order repaired and checked by reading, not compiled, not run on hardware
//
#include <hip/hip_runtime.h>

#define DEVINL __device__ __forceinline__

// Problem constants (reference: H=2048, NH=16, HD=128, B=2, S=2048)
constexpr int H  = 2048;
constexpr int NH = 16;
constexpr int HD = 128;
constexpr int S  = 2048;
constexpr int B  = 2;
constexpr int R  = B * S;   // 4096 total token rows

typedef __attribute__((ext_vector_type(16))) __bf16 bf16x16;
typedef __attribute__((ext_vector_type(8)))  float  f32x8;
typedef __attribute__((ext_vector_type(4)))  unsigned int u32x4;
typedef __attribute__((ext_vector_type(8)))  int  i32x8;
typedef __attribute__((ext_vector_type(4)))  int  i32x4;

union Frag16 {
  bf16x16 v;
  uint4 q[2];
  unsigned short u[16];
};
static_assert(sizeof(Frag16) == 32, "frag must be 32 bytes");

DEVINL unsigned short f2bf_bits(float f) {
  // fp32 -> bf16 round-to-nearest-even (bit pattern only)
  unsigned u = __float_as_uint(f);
  return (unsigned short)((u + 0x7FFFu + ((u >> 16) & 1u)) >> 16);
}

// A-fragment, 16x32 bf16 (ISA 7.12.2): lane l holds row M=l&15;
// elements 0..7 -> K = (l>>4)*8 + 0..7 ; elements 8..15 -> +16. Two b128 loads.
DEVINL bf16x16 load_frag_a(const unsigned short* base, int ld, int lane) {
  const unsigned short* p = base + (size_t)(lane & 15) * ld + ((lane >> 4) << 3);
  Frag16 f;
  f.q[0] = *reinterpret_cast<const uint4*>(p);
  f.q[1] = *reinterpret_cast<const uint4*>(p + 16);
  return f.v;
}

// B-fragment, 32x16 bf16: lane l holds column N=l&15 (a row of row-major W);
// elements 0..7 -> K = (l>>4)*8 + 0..7 ; elements 8..15 -> +16 (same as A).
DEVINL bf16x16 load_frag_b(const unsigned short* base, int ld, int lane) {
  const unsigned short* p = base + (size_t)(lane & 15) * ld + ((lane >> 4) << 3);
  Frag16 f;
  f.q[0] = *reinterpret_cast<const uint4*>(p);
  f.q[1] = *reinterpret_cast<const uint4*>(p + 16);
  return f.v;
}

DEVINL f32x8 wmma_bf16(bf16x16 a, bf16x16 b, f32x8 c) {
  return __builtin_amdgcn_wmma_f32_16x16x32_bf16(false, a, false, b, (short)0, c,
                                                 false, false);
}

// ---------------------------------------------------------------------------
// TDM: issue a 2-D tile DMA  global -> LDS  (bf16 elements, row-major tile).
// D# layout per CDNA5 ISA:
//   group0: count=1 | lds_addr | global_addr[56:0] | type=2
//   group1: data_size=2B, tile_dim0 (row elems), tile_dim1 (rows),
//           tensor_dim0_stride (row pitch in elems); huge tensor dims so the
//           OOB clamp never triggers (tiles divide the problem exactly).
// Tracked on TENSORcnt; caller waits with s_wait_tensorcnt.
// This toolchain exposes the 6-arg builtin:
//   (u32x4 g0, i32x8 g1, i32x4 g2, i32x4 g3, i32x8 extra, i32 cpol)
// ---------------------------------------------------------------------------
DEVINL void tdm_load_2d(const unsigned short* gsrc, unsigned lds_off_bytes,
                        int rows, int row_elems, int src_stride_elems) {
  const unsigned long long ga = (unsigned long long)(size_t)gsrc;
  u32x4 g0;
  g0[0] = 1u;                                        // count=1, user mode
  g0[1] = lds_off_bytes;                             // LDS byte address
  g0[2] = (unsigned)ga;                              // global addr [31:0]
  g0[3] = (unsigned)((ga >> 32) & 0x01FFFFFFu) | (2u << 30);  // [56:32] | type=2

  const unsigned td0 = 0x40000000u, td1 = 0x40000000u;   // huge tensor dims
  const unsigned long long st0 = (unsigned long long)src_stride_elems;
  i32x8 g1;
  g1[0] = (int)(1u << 16);                           // data_size=1 -> 2 bytes
  g1[1] = (int)((td0 & 0xFFFFu) << 16);              // tensor_dim0[15:0]
  g1[2] = (int)(((td0 >> 16) & 0xFFFFu) | ((td1 & 0xFFFFu) << 16));
  g1[3] = (int)(((td1 >> 16) & 0xFFFFu) |
                (((unsigned)row_elems & 0xFFFFu) << 16));   // tile_dim0
  g1[4] = (int)((unsigned)rows & 0xFFFFu);           // tile_dim1 ; tile_dim2=0
  g1[5] = (int)(unsigned)(st0 & 0xFFFFFFFFu);        // dim0 stride [31:0]
  g1[6] = (int)(unsigned)((st0 >> 32) & 0xFFFFu);    // dim0 stride [47:32]
  g1[7] = 0;                                         // dim1 stride unused (2-D)

  const i32x4 z4 = {0, 0, 0, 0};
  const i32x8 z8 = {0, 0, 0, 0, 0, 0, 0, 0};
  __builtin_amdgcn_tensor_load_to_lds(g0, g1, z4, z4, z8, 0);
}

DEVINL unsigned lds_offset_of(const void* p) {
  // Flat LDS aperture keeps the LDS byte offset in addr[31:0] (ISA §10.2).
  return (unsigned)(size_t)p;
}

// ---------------------------------------------------------------------------
// fp32 -> bf16 conversion
// ---------------------------------------------------------------------------
__global__ __launch_bounds__(256)
void cvt_f32_to_bf16(const float* __restrict__ in, unsigned short* __restrict__ out,
                     int n) {
  int i = blockIdx.x * 256 + threadIdx.x;
  if (i < n) out[i] = f2bf_bits(in[i]);
}

// ---------------------------------------------------------------------------
// GEMM: out[M x N] = A[M x K] * W[N x K]^T + bias  (torch y = x @ W.T + b)
// Block = 256 threads = 8 waves (2 in M x 4 in N). Block tile 64x256, K step 32.
// TDM double-buffers the 64x32 A tile and 256x32 W tile in LDS; wave 0 issues
// the DMA for step k+1 while all waves run WMMA on step k from LDS.
// EPI: 0 = bf16 out, 1 = relu+bf16 out, 2 = fp32 out,
//      3 = bf16 out transposed per head -> [B,NH,HD,S] (for V)
// ---------------------------------------------------------------------------
template <int EPI>
__global__ __launch_bounds__(256)
void gemm_bf16(const unsigned short* __restrict__ A,
               const unsigned short* __restrict__ W,
               const float* __restrict__ bias,
               void* __restrict__ out,
               int M, int N, int K) {
  constexpr int KS = 32;
  __shared__ __align__(16) unsigned short As[2][64 * KS];   //  2 x 4 KB
  __shared__ __align__(16) unsigned short Bs[2][256 * KS];  //  2 x 16 KB

  const int lane = threadIdx.x & 31;
  const int w    = threadIdx.x >> 5;
  const int wm = w >> 2;            // 0..1
  const int wn = w & 3;             // 0..3
  const int m0 = blockIdx.y * 64;
  const int n0 = blockIdx.x * 256;
  (void)M;

  const f32x8 fz = {0.f, 0.f, 0.f, 0.f, 0.f, 0.f, 0.f, 0.f};
  f32x8 acc[2][4];
#pragma unroll
  for (int i = 0; i < 2; i++)
#pragma unroll
    for (int j = 0; j < 4; j++) acc[i][j] = fz;

  const unsigned short* Abase = A + (size_t)m0 * K;
  const unsigned short* Wbase = W + (size_t)n0 * K;
  const int nk = K / KS;

  // Prologue: DMA tile 0 into buffer 0 (wave 0 only; TENSOR ops are per-wave)
  if (w == 0) {
    tdm_load_2d(Abase, lds_offset_of(&As[0][0]), 64, KS, K);
    tdm_load_2d(Wbase, lds_offset_of(&Bs[0][0]), 256, KS, K);
  }

  for (int it = 0; it < nk; it++) {
    const int buf = it & 1;
    if (w == 0) __builtin_amdgcn_s_wait_tensorcnt(0);  // tiles for `it` landed
    __syncthreads();  // LDS visible to all waves; prior compute (it-1) retired
    if (w == 0 && it + 1 < nk) {  // overlap DMA of it+1 with WMMA of it
      const int k1 = (it + 1) * KS;
      tdm_load_2d(Abase + k1, lds_offset_of(&As[buf ^ 1][0]), 64, KS, K);
      tdm_load_2d(Wbase + k1, lds_offset_of(&Bs[buf ^ 1][0]), 256, KS, K);
    }

    bf16x16 af[2], bfm[4];
#pragma unroll
    for (int i = 0; i < 2; i++)
      af[i] = load_frag_a(&As[buf][(wm * 32 + i * 16) * KS], KS, lane);
#pragma unroll
    for (int j = 0; j < 4; j++)
      bfm[j] = load_frag_b(&Bs[buf][(wn * 64 + j * 16) * KS], KS, lane);
#pragma unroll
    for (int i = 0; i < 2; i++)
#pragma unroll
      for (int j = 0; j < 4; j++)
        acc[i][j] = wmma_bf16(af[i], bfm[j], acc[i][j]);

    __syncthreads();  // all waves done reading `buf` before it is re-filled
  }

  // C/D layout: lane l -> col = l&15 ; VGPR r -> row = r + 8*(l>>4)
  const int halo = (lane >> 4) << 3;
  const int lc   = lane & 15;
  const int wm0  = m0 + wm * 32;
  const int wn0  = n0 + wn * 64;
#pragma unroll
  for (int i = 0; i < 2; i++) {
#pragma unroll
    for (int j = 0; j < 4; j++) {
      const int coln = wn0 + j * 16 + lc;
      const float bv = bias[coln];
#pragma unroll
      for (int r = 0; r < 8; r++) {
        const int rowm = wm0 + i * 16 + halo + r;
        float v = acc[i][j][r] + bv;
        if (EPI == 1) v = fmaxf(v, 0.f);
        if (EPI == 2) {
          ((float*)out)[(size_t)rowm * N + coln] = v;
        } else if (EPI == 3) {
          const int hh = coln >> 7;           // head  (HD = 128)
          const int dd = coln & (HD - 1);     // dim in head
          const int bb = rowm >> 11;          // batch (S = 2048)
          const int ss = rowm & (S - 1);      // seq pos
          ((unsigned short*)out)[(((size_t)(bb * NH + hh) * HD + dd) << 11) + ss] =
              f2bf_bits(v);
        } else {
          ((unsigned short*)out)[(size_t)rowm * N + coln] = f2bf_bits(v);
        }
      }
    }
  }
}

// ---------------------------------------------------------------------------
// Flash attention. One wave owns one (b, h, 16-row query tile).
// Streams 32-key blocks: S-tiles via WMMA (Q frags cached in regs), online
// softmax via wave32 xor-shuffles, P routed through LDS to A-fragment layout,
// P·V accumulated via WMMA into 16x128 fp32 accumulators.
// ---------------------------------------------------------------------------
__global__ __launch_bounds__(256)
void flash_attn(const unsigned short* __restrict__ Q,   // [B*S, H] bf16
                const unsigned short* __restrict__ Km,  // [B*S, H] bf16
                const unsigned short* __restrict__ Vt,  // [B*NH*HD, S] bf16
                unsigned short* __restrict__ O) {       // [B*S, H] bf16
  __shared__ __align__(16) unsigned short plds[8][16 * 32];

  const int lane = threadIdx.x & 31;
  const int w    = threadIdx.x >> 5;
  const int g    = blockIdx.x * 8 + w;   // global wave id
  const int qt   = g & (S / 16 - 1);     // query tile     (128)
  const int h    = (g >> 7) & (NH - 1);  // head           (16)
  const int b    = g >> 11;              // batch

  const unsigned short* qh = Q  + ((size_t)(b * S + qt * 16)) * H + h * HD;
  const unsigned short* kh = Km + ((size_t)(b * S)) * H + h * HD;
  const unsigned short* vh = Vt + ((size_t)((b * NH + h) * HD)) * S;
  unsigned short*       oh = O  + ((size_t)(b * S + qt * 16)) * H + h * HD;

  // Q A-fragments for the whole HD=128, cached in registers (4 x 16x32)
  bf16x16 qf[4];
#pragma unroll
  for (int kk = 0; kk < 4; kk++) qf[kk] = load_frag_a(qh + kk * 32, H, lane);

  const f32x8 fz = {0.f, 0.f, 0.f, 0.f, 0.f, 0.f, 0.f, 0.f};
  f32x8 o[8];
#pragma unroll
  for (int dt = 0; dt < 8; dt++) o[dt] = fz;
  float m8[8], l8[8];
#pragma unroll
  for (int r = 0; r < 8; r++) { m8[r] = -1e30f; l8[r] = 0.f; }

  const float sscale = 0.08838834764831845f;  // 1/sqrt(128)
  const int lc = lane & 15;
  const int halo = (lane >> 4) << 3;

  for (int kb = 0; kb < S / 32; kb++) {
    const unsigned short* kp0 = kh + (size_t)(kb * 32) * H;
    f32x8 s0 = fz, s1 = fz;
#pragma unroll
    for (int kk = 0; kk < 4; kk++) {
      bf16x16 kb0 = load_frag_b(kp0 + kk * 32, H, lane);                   // keys +0..15
      bf16x16 kb1 = load_frag_b(kp0 + (size_t)16 * H + kk * 32, H, lane);  // keys +16..31
      s0 = wmma_bf16(qf[kk], kb0, s0);
      s1 = wmma_bf16(qf[kk], kb1, s1);
    }

    // Online softmax over this 16x32 score block
#pragma unroll
    for (int r = 0; r < 8; r++) {
      float a0 = s0[r] * sscale;
      float a1 = s1[r] * sscale;
      float mx = fmaxf(a0, a1);
#pragma unroll
      for (int dd = 1; dd < 16; dd <<= 1) mx = fmaxf(mx, __shfl_xor(mx, dd, 32));
      const float mnew = fmaxf(m8[r], mx);
      const float corr = __expf(m8[r] - mnew);
      const float p0 = __expf(a0 - mnew);
      const float p1 = __expf(a1 - mnew);
      float rs = p0 + p1;
#pragma unroll
      for (int dd = 1; dd < 16; dd <<= 1) rs += __shfl_xor(rs, dd, 32);
      l8[r] = l8[r] * corr + rs;
      m8[r] = mnew;
#pragma unroll
      for (int dt = 0; dt < 8; dt++) o[dt][r] *= corr;
      const int prow = r + halo;                       // C layout row
      plds[w][prow * 32 + lc]      = f2bf_bits(p0);
      plds[w][prow * 32 + 16 + lc] = f2bf_bits(p1);
    }

    // Reload P as a 16x32 A-fragment (transpose via LDS; wave-private, LDS is
    // in-order per wave so no barrier needed)
    Frag16 pf;
    {
      const unsigned short* pp = &plds[w][(size_t)lc * 32 + halo];
      pf.q[0] = *reinterpret_cast<const uint4*>(pp);
      pf.q[1] = *reinterpret_cast<const uint4*>(pp + 16);
    }

    // P (16x32) x V-block (32x128): 8 WMMAs from transposed-V layout
#pragma unroll
    for (int dt = 0; dt < 8; dt++) {
      bf16x16 vb = load_frag_b(vh + (size_t)(dt * 16) * S + kb * 32, S, lane);
      o[dt] = wmma_bf16(pf.v, vb, o[dt]);
    }
  }

  // Finalize: divide by softmax denominator, emit bf16 context
#pragma unroll
  for (int r = 0; r < 8; r++) {
    const float inv = 1.f / l8[r];
    const int rowq = r + halo;
#pragma unroll
    for (int dt = 0; dt < 8; dt++)
      oh[(size_t)rowq * H + dt * 16 + lc] = f2bf_bits(o[dt][r] * inv);
  }
}

// ---------------------------------------------------------------------------
// LayerNorm(y + residual) * g + b ; fp32 out (+ optional bf16 copy)
// One 256-thread block per row of H=2048 (8 elements per thread).
// ---------------------------------------------------------------------------
__global__ __launch_bounds__(256)
void ln_kernel(const float* __restrict__ y, const float* __restrict__ res,
               const float* __restrict__ gamma, const float* __restrict__ beta,
               float* __restrict__ out_f32, unsigned short* __restrict__ out_bf) {
  __shared__ float s_sum[8], s_sq[8];
  __shared__ float s_mean, s_rstd;

  const int row  = blockIdx.x;
  const int tid  = threadIdx.x;
  const int lane = tid & 31;
  const int wid  = tid >> 5;
  const size_t base = (size_t)row * H;

  float v[8];
  float lsum = 0.f, lsq = 0.f;
#pragma unroll
  for (int i = 0; i < 8; i++) {
    const int idx = i * 256 + tid;
    const float t = y[base + idx] + res[base + idx];
    v[i] = t;
    lsum += t;
    lsq  += t * t;
  }
#pragma unroll
  for (int dd = 1; dd < 32; dd <<= 1) {
    lsum += __shfl_xor(lsum, dd, 32);
    lsq  += __shfl_xor(lsq,  dd, 32);
  }
  if (lane == 0) { s_sum[wid] = lsum; s_sq[wid] = lsq; }
  __syncthreads();
  if (tid == 0) {
    float ts = 0.f, tq = 0.f;
#pragma unroll
    for (int i = 0; i < 8; i++) { ts += s_sum[i]; tq += s_sq[i]; }
    const float mean = ts * (1.f / H);
    const float var  = tq * (1.f / H) - mean * mean;
    s_mean = mean;
    s_rstd = rsqrtf(var + 1e-5f);
  }
  __syncthreads();
  const float mean = s_mean, rstd = s_rstd;
#pragma unroll
  for (int i = 0; i < 8; i++) {
    const int idx = i * 256 + tid;
    const float o = (v[i] - mean) * rstd * gamma[idx] + beta[idx];
    out_f32[base + idx] = o;
    if (out_bf) out_bf[base + idx] = f2bf_bits(o);
  }
}

// ---------------------------------------------------------------------------
// Host orchestration
// ---------------------------------------------------------------------------
extern "C" void kernel_launch(void* const* d_in, const int* in_sizes, int n_in,
                              void* d_out, int out_size, void* d_ws, size_t ws_size,
                              hipStream_t stream) {
  const float* x   = (const float*)d_in[0];
  const float* Wq  = (const float*)d_in[1];
  const float* bq  = (const float*)d_in[2];
  const float* Wk  = (const float*)d_in[3];
  const float* bk  = (const float*)d_in[4];
  const float* Wv  = (const float*)d_in[5];
  const float* bv  = (const float*)d_in[6];
  const float* Wo  = (const float*)d_in[7];
  const float* bo  = (const float*)d_in[8];
  const float* lng = (const float*)d_in[9];
  const float* lnb = (const float*)d_in[10];
  const float* W1  = (const float*)d_in[11];
  const float* b1  = (const float*)d_in[12];
  const float* W2  = (const float*)d_in[13];
  const float* b2  = (const float*)d_in[14];
  const float* fg  = (const float*)d_in[15];
  const float* fb  = (const float*)d_in[16];
  float* out = (float*)d_out;

  char* ws = (char*)d_ws;
  const size_t WB = (size_t)H * H * sizeof(unsigned short);  // 8 MB per weight
  const size_t AB = (size_t)R * H * sizeof(unsigned short);  // 16 MB activation
  const size_t FB = (size_t)R * H * sizeof(float);           // 32 MB fp32 act

  unsigned short* wq_bf  = (unsigned short*)(ws + 0 * WB);
  unsigned short* wk_bf  = (unsigned short*)(ws + 1 * WB);
  unsigned short* wv_bf  = (unsigned short*)(ws + 2 * WB);
  unsigned short* wo_bf  = (unsigned short*)(ws + 3 * WB);
  unsigned short* w1_bf  = (unsigned short*)(ws + 4 * WB);
  unsigned short* w2_bf  = (unsigned short*)(ws + 5 * WB);
  char* act = ws + 6 * WB;
  unsigned short* x_bf   = (unsigned short*)(act + 0 * AB);
  unsigned short* q_bf   = (unsigned short*)(act + 1 * AB);
  unsigned short* k_bf   = (unsigned short*)(act + 2 * AB);
  unsigned short* vt_bf  = (unsigned short*)(act + 3 * AB);
  float*          y_f32  = (float*)(act + 4 * AB);
  float*          ln1_f  = (float*)(act + 4 * AB + FB);
  unsigned short* ln1_bf = (unsigned short*)(act + 4 * AB + 2 * FB);
  // Reuse: x_bf slot -> attention context, q_bf slot -> FFN hidden
  unsigned short* ctx_bf = x_bf;
  unsigned short* h_bf   = q_bf;

  const int nw = H * H;     // weight elements
  const int na = R * H;     // activation elements
  cvt_f32_to_bf16<<<dim3((nw + 255) / 256), dim3(256), 0, stream>>>(Wq, wq_bf, nw);
  cvt_f32_to_bf16<<<dim3((nw + 255) / 256), dim3(256), 0, stream>>>(Wk, wk_bf, nw);
  cvt_f32_to_bf16<<<dim3((nw + 255) / 256), dim3(256), 0, stream>>>(Wv, wv_bf, nw);
  cvt_f32_to_bf16<<<dim3((nw + 255) / 256), dim3(256), 0, stream>>>(Wo, wo_bf, nw);
  cvt_f32_to_bf16<<<dim3((nw + 255) / 256), dim3(256), 0, stream>>>(W1, w1_bf, nw);
  cvt_f32_to_bf16<<<dim3((nw + 255) / 256), dim3(256), 0, stream>>>(W2, w2_bf, nw);
  cvt_f32_to_bf16<<<dim3((na + 255) / 256), dim3(256), 0, stream>>>(x, x_bf, na);

  const dim3 gblk(256);
  const dim3 ggrid(H / 256, R / 64);  // (8, 64)

  // Q / K / V projections
  gemm_bf16<0><<<ggrid, gblk, 0, stream>>>(x_bf, wq_bf, bq, q_bf, R, H, H);
  gemm_bf16<0><<<ggrid, gblk, 0, stream>>>(x_bf, wk_bf, bk, k_bf, R, H, H);
  gemm_bf16<3><<<ggrid, gblk, 0, stream>>>(x_bf, wv_bf, bv, vt_bf, R, H, H);

  // Flash attention: B*NH*(S/16) = 4096 waves -> 512 blocks of 8 waves
  flash_attn<<<dim3((B * NH * (S / 16)) / 8), gblk, 0, stream>>>(q_bf, k_bf,
                                                                 vt_bf, ctx_bf);

  // Output projection + LN1 (residual = x)
  gemm_bf16<2><<<ggrid, gblk, 0, stream>>>(ctx_bf, wo_bf, bo, y_f32, R, H, H);
  ln_kernel<<<dim3(R), gblk, 0, stream>>>(y_f32, x, lng, lnb, ln1_f, ln1_bf);

  // FFN + LN2 (residual = ln1)
  gemm_bf16<1><<<ggrid, gblk, 0, stream>>>(ln1_bf, w1_bf, b1, h_bf, R, H, H);
  gemm_bf16<2><<<ggrid, gblk, 0, stream>>>(h_bf, w2_bf, b2, y_f32, R, H, H);
  ln_kernel<<<dim3(R), gblk, 0, stream>>>(y_f32, ln1_f, fg, fb, out, nullptr);
}
